// LightGCNPlus3_3539053052414
// MI455X (gfx1250) — matrix operand order repaired and checked by reading, not compiled, not run on hardware
//
#include <hip/hip_runtime.h>
#include <hip/hip_bf16.h>

typedef __attribute__((ext_vector_type(16))) _Float16 v16h;
typedef __attribute__((ext_vector_type(8)))  float    v8f;

#define D 64          // embedding dim
#define F 256         // feature dim
#define INV25 0.04f   // 1/(L+1)^2, L=4

// ---------------- utility ----------------
__global__ void lgcn_fill_f32(float* __restrict__ p, float v, int n) {
    int i = blockIdx.x * blockDim.x + threadIdx.x;
    if (i < n) p[i] = v;
}

// deg[c] += 1 for every edge endpoint c (symmetric-norm degree, computed on col like the reference)
__global__ void lgcn_degree(const int* __restrict__ col, float* __restrict__ deg, int E) {
    int e = blockIdx.x * blockDim.x + threadIdx.x;
    if (e < E) unsafeAtomicAdd(&deg[col[e]], 1.0f);
}

__global__ void lgcn_dinv(const float* __restrict__ deg, float* __restrict__ dinv, int n) {
    int i = blockIdx.x * blockDim.x + threadIdx.x;
    if (i < n) {
        float d = deg[i];
        dinv[i] = (d > 0.0f) ? __frsqrt_rn(fmaxf(d, 1.0f)) : 0.0f;
    }
}

// x = concat(emb_users, emb_items); acc(d_out) = x
__global__ void lgcn_init(const float* __restrict__ eu, const float* __restrict__ ei,
                          float* __restrict__ x, float* __restrict__ acc,
                          int nu_elems, int total_elems) {
    int i = blockIdx.x * blockDim.x + threadIdx.x;
    if (i < total_elems) {
        float v = (i < nu_elems) ? eu[i] : ei[i - nu_elems];
        x[i] = v;
        acc[i] = v;
    }
}

// one propagation layer: xout[col] += xin[row] * dinv[row]*dinv[col]
// thread handles one (edge, 4-dim quad): float4 gather + 4 HW float atomics
__global__ void lgcn_scatter(const int* __restrict__ row, const int* __restrict__ col,
                             const float* __restrict__ dinv,
                             const float* __restrict__ xin, float* __restrict__ xout, int E) {
    int t = blockIdx.x * blockDim.x + threadIdx.x;
    int e = t >> 4;
    if (e >= E) return;
    int q = (t & 15) << 2;
    int r = row[e], c = col[e];
    float nrm = dinv[r] * dinv[c];
    const float4 v = *(const float4*)(xin + (size_t)r * D + q);
    float* dst = xout + (size_t)c * D + q;
    unsafeAtomicAdd(dst + 0, v.x * nrm);
    unsafeAtomicAdd(dst + 1, v.y * nrm);
    unsafeAtomicAdd(dst + 2, v.z * nrm);
    unsafeAtomicAdd(dst + 3, v.w * nrm);
}

__global__ void lgcn_accadd(float* __restrict__ acc, const float* __restrict__ x, int n) {
    int i = blockIdx.x * blockDim.x + threadIdx.x;
    if (i < n) acc[i] += x[i];
}

// Pack W (EMB_DIM=64 x F=256, row-major f32, out = feat @ W^T so B[k][n] = W[n][k])
// into f16 B-fragments matching v_wmma_f32_16x16x32_f16's B layout:
//   fragment (kt, nt) is 32(K) x 16(N); lane l holds N = nt*16 + (l&15),
//   halves 0..7 are K = kt*32 + 8*(l>>4) + 0..7, halves 8..15 are K = kt*32 + 16 + 8*(l>>4) + 0..7.
// Linear layout: frag[(kt*4+nt)] [lane] [16 halves]  (32B per lane -> v16h load)
__global__ void lgcn_pack_w(const float* __restrict__ Wu, const float* __restrict__ Wi,
                            _Float16* __restrict__ Bu, _Float16* __restrict__ Bi) {
    int t = blockIdx.x * blockDim.x + threadIdx.x;   // 2 * 8*4*32*16 = 32768 threads
    if (t >= 2 * 8 * 4 * 32 * 16) return;
    int mat = t >> 14;
    int r   = t & 16383;
    int kt  = r >> 11;
    int r2  = r & 2047;
    int nt  = r2 >> 9;
    int r3  = r2 & 511;
    int lane = r3 >> 4;
    int hi   = r3 & 15;
    int k = kt * 32 + ((hi >> 3) << 4) + ((lane >> 4) << 3) + (hi & 7);
    int n = nt * 16 + (lane & 15);
    const float* W = mat ? Wi : Wu;
    _Float16*    B = mat ? Bi : Bu;
    B[r] = (_Float16)W[(size_t)n * F + k];
}

// One wave computes a 16(M) x 64(N) tile of  out = acc*1/25 + feat @ W^T.
// outacc points at the first output row for this matrix (d_out, holding acc).
__global__ void __launch_bounds__(256)
lgcn_proj_wmma(const float* __restrict__ feat, const _Float16* __restrict__ Bpack,
               float* __restrict__ outacc, int rows) {
    const int lane = threadIdx.x & 31;
    const int wave = threadIdx.x >> 5;
    const int tile = blockIdx.x * 8 + wave;
    const int M0 = tile * 16;
    if (M0 >= rows) return;

    const int lrow = lane & 15;   // A: M row / C: N col
    const int lhi  = lane >> 4;   // A: K-half select / C: M+8 select

    const v16h* Bf = (const v16h*)Bpack;

    v8f c[4] = {};
    const float* arow = feat + (size_t)(M0 + lrow) * F + lhi * 8;

#pragma unroll
    for (int kt = 0; kt < 8; ++kt) {
        // A fragment: 16x32 f16; lane holds M=lrow, halves 0..7 = K kt*32+lhi*8+0..7,
        // halves 8..15 = K kt*32+16+lhi*8+0..7  (per ISA 16-bit A layout)
        const float* ap = arow + kt * 32;
        float fa[16];
        *(float4*)(&fa[0])  = *(const float4*)(ap);
        *(float4*)(&fa[4])  = *(const float4*)(ap + 4);
        *(float4*)(&fa[8])  = *(const float4*)(ap + 16);
        *(float4*)(&fa[12]) = *(const float4*)(ap + 20);
        v16h a;
#pragma unroll
        for (int i = 0; i < 16; ++i) a[i] = (_Float16)fa[i];

#pragma unroll
        for (int nt = 0; nt < 4; ++nt) {
            v16h b = Bf[(kt * 4 + nt) * 32 + lane];
            c[nt] = __builtin_amdgcn_wmma_f32_16x16x32_f16(
                false, a, false, b, (short)0, c[nt], false, false);
        }
    }

    // C layout: lane l, vgpr v -> M = v + 8*(l>>4), N = l&15. Fuse acc/25 epilogue.
#pragma unroll
    for (int nt = 0; nt < 4; ++nt) {
#pragma unroll
        for (int v = 0; v < 8; ++v) {
            size_t idx = ((size_t)M0 + v + 8 * lhi) * D + nt * 16 + lrow;
            outacc[idx] = outacc[idx] * INV25 + c[nt][v];
        }
    }
}

extern "C" void kernel_launch(void* const* d_in, const int* in_sizes, int n_in,
                              void* d_out, int out_size, void* d_ws, size_t ws_size,
                              hipStream_t stream) {
    const int E  = in_sizes[0] / 2;
    const int NU = in_sizes[1] / D;
    const int NI = in_sizes[2] / D;
    const int N  = NU + NI;
    const int NE = N * D;            // node-matrix element count (9.6M, fits int)

    const int* edge = (const int*)d_in[0];
    const int* row  = edge;
    const int* col  = edge + E;
    const float* eu = (const float*)d_in[1];
    const float* ei = (const float*)d_in[2];
    const float* fu = (const float*)d_in[3];
    const float* fi = (const float*)d_in[4];
    const float* Wu = (const float*)d_in[5];
    const float* Wi = (const float*)d_in[6];
    float* out = (float*)d_out;      // also serves as the layer accumulator

    // carve workspace (256B aligned)
    char* w = (char*)d_ws;
    auto carve = [&](size_t bytes) { void* p = (void*)w; w += (bytes + 255) & ~(size_t)255; return p; };
    float*    xa   = (float*)carve((size_t)NE * 4);
    float*    xb   = (float*)carve((size_t)NE * 4);
    float*    deg  = (float*)carve((size_t)N * 4);
    float*    dinv = (float*)carve((size_t)N * 4);
    _Float16* Bu   = (_Float16*)carve(16384 * 2);
    _Float16* Bi   = (_Float16*)carve(16384 * 2);

    const int T = 256;
    // degree + dinv
    lgcn_fill_f32<<<(N + T - 1) / T, T, 0, stream>>>(deg, 0.0f, N);
    lgcn_degree<<<(E + T - 1) / T, T, 0, stream>>>(col, deg, E);
    lgcn_dinv<<<(N + T - 1) / T, T, 0, stream>>>(deg, dinv, N);

    // x0 and acc init
    lgcn_init<<<(NE + T - 1) / T, T, 0, stream>>>(eu, ei, xa, out, NU * D, NE);

    // 4 propagation layers (ping-pong xa/xb, accumulate into out)
    float* xin = xa;
    float* xout = xb;
    for (int layer = 0; layer < 4; ++layer) {
        lgcn_fill_f32<<<(NE + T - 1) / T, T, 0, stream>>>(xout, 0.0f, NE);
        int st = E * 16;
        lgcn_scatter<<<(st + T - 1) / T, T, 0, stream>>>(row, col, dinv, xin, xout, E);
        lgcn_accadd<<<(NE + T - 1) / T, T, 0, stream>>>(out, xout, NE);
        float* tmp = xin; xin = xout; xout = tmp;
    }

    // pack weights into WMMA B-fragment layout
    lgcn_pack_w<<<(32768 + T - 1) / T, T, 0, stream>>>(Wu, Wi, Bu, Bi);

    // fused projection + acc/25 epilogue (users then items)
    int ublocks = (NU / 16 + 7) / 8;
    int iblocks = (NI / 16 + 7) / 8;
    lgcn_proj_wmma<<<ublocks, T, 0, stream>>>(fu, Bu, out, NU);
    lgcn_proj_wmma<<<iblocks, T, 0, stream>>>(fi, Bi, out + (size_t)NU * D, NI);
}
